// IdpGANGenerator_30760555774599
// MI455X (gfx1250) — compile-verified, matrix operand order checked by reading, not run on hardware
//
#include <hip/hip_runtime.h>

typedef __attribute__((ext_vector_type(16))) _Float16 v16h;
typedef __attribute__((ext_vector_type(8)))  float    v8f;

#define NBATCH 8
#define SEQL   512
#define EMBEDD 256
#define NHEAD  16
#define HEADD  16
#define DFFD   1024
#define E1DD   32
#define PDIMD  64
#define PMAXLD 24
#define NLAYER 6
#define MROWS  (NBATCH * SEQL)   // 4096

// ---------------- WMMA fragment helpers (wave32, 16x16x32 f16) ----------------

__device__ __forceinline__ v8f wmma_f16(v16h a, v16h b, v8f c) {
  // D = A(16x32 f16) x B(32x16 f16) + C(16x16 f32)
  return __builtin_amdgcn_wmma_f32_16x16x32_f16(false, a, false, b, (short)0, c,
                                                false, false);
}

// A/B fragment from row-major f32 matrix, full 32-wide K tile, via b128 loads.
// Lane layout: row = lane&15 ; lane half selects K 0-7/16-23 vs 8-15/24-31.
__device__ __forceinline__ v16h frag_f32(const float* __restrict__ base, int ld,
                                         int lane, int k0) {
  const int row = lane & 15;
  const int kb  = k0 + ((lane >> 4) << 3);
  const float* p = base + (size_t)row * ld + kb;
  const float4 q0 = ((const float4*)p)[0];        // K = kb..kb+3
  const float4 q1 = ((const float4*)p)[1];        // K = kb+4..kb+7
  const float4 q2 = ((const float4*)(p + 16))[0]; // K = kb+16..kb+19
  const float4 q3 = ((const float4*)(p + 16))[1]; // K = kb+20..kb+23
  v16h f;
  f[0]  = (_Float16)q0.x; f[1]  = (_Float16)q0.y; f[2]  = (_Float16)q0.z; f[3]  = (_Float16)q0.w;
  f[4]  = (_Float16)q1.x; f[5]  = (_Float16)q1.y; f[6]  = (_Float16)q1.z; f[7]  = (_Float16)q1.w;
  f[8]  = (_Float16)q2.x; f[9]  = (_Float16)q2.y; f[10] = (_Float16)q2.z; f[11] = (_Float16)q2.w;
  f[12] = (_Float16)q3.x; f[13] = (_Float16)q3.y; f[14] = (_Float16)q3.z; f[15] = (_Float16)q3.w;
  return f;
}

// Fragment where only K=0..15 is valid (head-dim 16 padded to 32 with zeros).
__device__ __forceinline__ v16h frag_f32_k16(const float* __restrict__ base,
                                             int ld, int lane) {
  const int row = lane & 15;
  const int kb  = (lane >> 4) << 3;               // 0 or 8
  const float* p = base + (size_t)row * ld + kb;
  const float4 q0 = ((const float4*)p)[0];
  const float4 q1 = ((const float4*)p)[1];
  v16h f;
  f[0] = (_Float16)q0.x; f[1] = (_Float16)q0.y; f[2] = (_Float16)q0.z; f[3] = (_Float16)q0.w;
  f[4] = (_Float16)q1.x; f[5] = (_Float16)q1.y; f[6] = (_Float16)q1.z; f[7] = (_Float16)q1.w;
#pragma unroll
  for (int e = 8; e < 16; ++e) f[e] = (_Float16)0.0f;   // K >= 16 zero padding
  return f;
}

// ---------------- GEMM: C = act(A @ W^T * scale + bias) ------------------------
// A: (M x K) row-major f32, W: (Nout x K) row-major f32.
// Wave tile 32x32 (4 WMMA accumulators), block = 8 waves as 2x4 -> 64 x 128.
// M % 64 == 0, Nout % 128 == 0, K % 32 == 0, and M rows are NBATCH x SEQL.
// TRANSOUT=false: C row-major (M x Nout).
// TRANSOUT=true:  C written as (NBATCH, Nout, SEQL) -- i.e. per-batch transposed,
//                 so the attention kernel can read V^T tiles with b128 loads.
//                 Fragment rows become contiguous -> float4 stores.

template <bool RELU, bool HASBIAS, bool TRANSOUT>
__global__ __launch_bounds__(256) void k_gemm(
    const float* __restrict__ A, const float* __restrict__ Wt,
    const float* __restrict__ bias, float* __restrict__ C,
    int M, int Nout, int K, float scale) {
  const int lane = threadIdx.x & 31;
  const int wave = threadIdx.x >> 5;
  const int m0 = blockIdx.x * 64 + (wave >> 2) * 32;
  const int n0 = blockIdx.y * 128 + (wave & 3) * 32;

  v8f acc00 = {}, acc01 = {}, acc10 = {}, acc11 = {};
  const float* A0 = A  + (size_t)m0 * K;
  const float* A1 = A0 + (size_t)16 * K;
  const float* B0 = Wt + (size_t)n0 * K;
  const float* B1 = B0 + (size_t)16 * K;

  for (int k0 = 0; k0 < K; k0 += 32) {
    if (k0 + 32 < K) {  // pull the next K-slab toward the caches
      const size_t ro = (size_t)(lane & 15) * K + k0 + 32;
      __builtin_prefetch(A0 + ro, 0, 1);
      __builtin_prefetch(A1 + ro, 0, 1);
      __builtin_prefetch(B0 + ro, 0, 1);
      __builtin_prefetch(B1 + ro, 0, 1);
    }
    v16h a0 = frag_f32(A0, K, lane, k0);
    v16h a1 = frag_f32(A1, K, lane, k0);
    v16h b0 = frag_f32(B0, K, lane, k0);
    v16h b1 = frag_f32(B1, K, lane, k0);
    acc00 = wmma_f16(a0, b0, acc00);
    acc01 = wmma_f16(a0, b1, acc01);
    acc10 = wmma_f16(a1, b0, acc10);
    acc11 = wmma_f16(a1, b1, acc11);
  }

  const int half = lane >> 4, n = lane & 15;
  const float bc0 = HASBIAS ? bias[n0 + n]      : 0.0f;   // hoisted: no per-row branch
  const float bc1 = HASBIAS ? bias[n0 + 16 + n] : 0.0f;
  float v00[8], v01[8], v10[8], v11[8];
#pragma unroll
  for (int r = 0; r < 8; ++r) {
    v00[r] = acc00[r] * scale + bc0;
    v01[r] = acc01[r] * scale + bc1;
    v10[r] = acc10[r] * scale + bc0;
    v11[r] = acc11[r] * scale + bc1;
    if (RELU) {
      v00[r] = fmaxf(v00[r], 0.0f); v01[r] = fmaxf(v01[r], 0.0f);
      v10[r] = fmaxf(v10[r], 0.0f); v11[r] = fmaxf(v11[r], 0.0f);
    }
  }

  if (TRANSOUT) {
    // C[(b*Nout + col)*SEQL + j] ; fragment rows are contiguous j -> b128 stores
    const int b_    = m0 >> 9;            // uniform: 32-row tile never crosses batch
    const int jbase = (m0 & 511) + 8 * half;
    float* p00 = C + ((size_t)(b_ * Nout + n0 + n)      * SEQL) + jbase;
    float* p01 = C + ((size_t)(b_ * Nout + n0 + 16 + n) * SEQL) + jbase;
    ((float4*)p00)[0] = make_float4(v00[0], v00[1], v00[2], v00[3]);
    ((float4*)p00)[1] = make_float4(v00[4], v00[5], v00[6], v00[7]);
    ((float4*)p01)[0] = make_float4(v01[0], v01[1], v01[2], v01[3]);
    ((float4*)p01)[1] = make_float4(v01[4], v01[5], v01[6], v01[7]);
    ((float4*)(p00 + 16))[0] = make_float4(v10[0], v10[1], v10[2], v10[3]);
    ((float4*)(p00 + 16))[1] = make_float4(v10[4], v10[5], v10[6], v10[7]);
    ((float4*)(p01 + 16))[0] = make_float4(v11[0], v11[1], v11[2], v11[3]);
    ((float4*)(p01 + 16))[1] = make_float4(v11[4], v11[5], v11[6], v11[7]);
  } else {
    float* c0 = C + (size_t)(m0 + 8 * half) * Nout + n0 + n;        // rows m0+8h..
    float* c1 = C + (size_t)(m0 + 16 + 8 * half) * Nout + n0 + n;   // rows m0+16+8h..
#pragma unroll
    for (int r = 0; r < 8; ++r) {
      c0[(size_t)r * Nout]      = v00[r];
      c0[(size_t)r * Nout + 16] = v01[r];
      c1[(size_t)r * Nout]      = v10[r];
      c1[(size_t)r * Nout + 16] = v11[r];
    }
  }
}

// ---------------- fused flash attention, transposed-score layout ---------------
// One wave per (batch, head, 16-query tile). We compute S^T = K_tile @ Q^T so
// each lane owns a single query row: softmax over keys reduces in registers
// (one shfl_xor(16) for max, one for sum per 32-key tile), and exp(S^T) lands
// directly in the B-fragment layout required by O^T = V^T @ P -- no LDS staging.
// V arrives pre-transposed as (NBATCH, EMBEDD, SEQL) so V^T tiles load as b128.

__global__ __launch_bounds__(256) void k_attn(
    const float* __restrict__ q, const float* __restrict__ kbuf,
    const float* __restrict__ vT, const float* __restrict__ bias,
    float* __restrict__ obuf) {
  const int lane = threadIdx.x & 31;
  const int wave = threadIdx.x >> 5;
  const int gw = blockIdx.x * 8 + wave;         // 0..4095
  const int n_ = gw >> 9;                       // batch
  const int h_ = (gw >> 5) & 15;                // head
  const int i0 = (gw & 31) << 4;                // query row tile
  const int half = lane >> 4, nl = lane & 15;
  const int i = i0 + nl;                        // query row owned by this lane

  // Q as the B operand (K-dim = head-dim 16, zero-padded to 32); loaded once.
  v16h bQ = frag_f32_k16(q + ((size_t)(n_ * SEQL + i0)) * EMBEDD + h_ * HEADD,
                         EMBEDD, lane);
  v8f o = {};                                   // O^T[d = r+8*half][i]
  float m = -1e30f, s = 0.0f;
  const float* biasrow = bias + ((size_t)h_ * SEQL + i) * SEQL;
  const float* vbase   = vT + ((size_t)(n_ * EMBEDD + h_ * HEADD)) * SEQL;

  for (int j0 = 0; j0 < SEQL; j0 += 32) {
    const float* kp = kbuf + ((size_t)(n_ * SEQL + j0)) * EMBEDD + h_ * HEADD;
    v16h aK0 = frag_f32_k16(kp, EMBEDD, lane);
    v16h aK1 = frag_f32_k16(kp + (size_t)16 * EMBEDD, EMBEDD, lane);
    v8f s0 = {}, s1 = {};
    s0 = wmma_f16(aK0, bQ, s0);                 // S^T[j0    + r+8h][i]
    s1 = wmma_f16(aK1, bQ, s1);                 // S^T[j0+16 + r+8h][i]

    // bias[j] is contiguous per lane in this layout: 4 x b128 loads
    const float* brow = biasrow + j0 + 8 * half;
    const float4 ba = ((const float4*)brow)[0];
    const float4 bb = ((const float4*)brow)[1];
    const float4 bc = ((const float4*)(brow + 16))[0];
    const float4 bd = ((const float4*)(brow + 16))[1];
    float x0[8], x1[8];
    x0[0] = s0[0] + ba.x; x0[1] = s0[1] + ba.y; x0[2] = s0[2] + ba.z; x0[3] = s0[3] + ba.w;
    x0[4] = s0[4] + bb.x; x0[5] = s0[5] + bb.y; x0[6] = s0[6] + bb.z; x0[7] = s0[7] + bb.w;
    x1[0] = s1[0] + bc.x; x1[1] = s1[1] + bc.y; x1[2] = s1[2] + bc.z; x1[3] = s1[3] + bc.w;
    x1[4] = s1[4] + bd.x; x1[5] = s1[5] + bd.y; x1[6] = s1[6] + bd.z; x1[7] = s1[7] + bd.w;

    float lm = x0[0];
#pragma unroll
    for (int r = 1; r < 8; ++r) lm = fmaxf(lm, x0[r]);
#pragma unroll
    for (int r = 0; r < 8; ++r) lm = fmaxf(lm, x1[r]);
    lm = fmaxf(lm, __shfl_xor(lm, 16, 32));     // merge the two lane halves
    const float mn   = fmaxf(m, lm);
    const float corr = __expf(m - mn);

    v16h pf;                                    // exp(S^T) is already B-layout
    float ps = 0.0f;
#pragma unroll
    for (int r = 0; r < 8; ++r) {
      const float p0 = __expf(x0[r] - mn);
      const float p1 = __expf(x1[r] - mn);
      ps += p0 + p1;
      pf[r]     = (_Float16)p0;
      pf[r + 8] = (_Float16)p1;
    }
    ps += __shfl_xor(ps, 16, 32);               // full 32-key partial sum
    s = s * corr + ps;
    m = mn;
#pragma unroll
    for (int r = 0; r < 8; ++r) o[r] *= corr;

    v16h aV = frag_f32(vbase, SEQL, lane, j0);  // V^T tile: rows=d, K=j, b128 loads
    o = wmma_f16(aV, pf, o);                    // O^T += V^T @ P
  }

  const float inv = 1.0f / s;
  float* op = obuf + ((size_t)(n_ * SEQL + i)) * EMBEDD + h_ * HEADD + 8 * half;
  const float4 o0 = make_float4(o[0] * inv, o[1] * inv, o[2] * inv, o[3] * inv);
  const float4 o1 = make_float4(o[4] * inv, o[5] * inv, o[6] * inv, o[7] * inv);
  ((float4*)op)[0] = o0;
  ((float4*)op)[1] = o1;
}

// ---------------- small supporting kernels ------------------------------------

__global__ void k_tz(const float* __restrict__ z, float* __restrict__ hin) {
  int idx = blockIdx.x * blockDim.x + threadIdx.x;   // N*L*64
  if (idx >= MROWS * 64) return;
  const int c = idx & 63, i = (idx >> 6) & 511, n = idx >> 15;
  hin[idx] = z[((size_t)n * 64 + c) * SEQL + i];
}

__global__ void k_eaa(const float* __restrict__ x, const float* __restrict__ aa,
                      float* __restrict__ eaa) {
  int idx = blockIdx.x * blockDim.x + threadIdx.x;   // N*L
  if (idx >= MROWS) return;
  const int n = idx >> 9, i = idx & 511;
  int best = 0; float bv = x[((size_t)n * 20) * SEQL + i];
  for (int a = 1; a < 20; ++a) {
    float xv = x[((size_t)n * 20 + a) * SEQL + i];
    if (xv > bv) { bv = xv; best = a; }
  }
#pragma unroll
  for (int e = 0; e < E1DD; ++e)
    eaa[(size_t)idx * E1DD + e] = aa[best * E1DD + e];
}

__global__ __launch_bounds__(256) void k_bias2d(
    const float* __restrict__ pos_emb, const float* __restrict__ W2d,
    const float* __restrict__ b2d, float* __restrict__ bias) {
  __shared__ float w2s[PDIMD];
  const int i = blockIdx.x, h_ = blockIdx.y, t = threadIdx.x;
  if (t < PDIMD) w2s[t] = W2d[h_ * PDIMD + t];
  __syncthreads();
  const float bb = b2d[h_];
  for (int j = t; j < SEQL; j += 256) {
    int rel = j - i;
    rel = rel < -PMAXLD ? -PMAXLD : (rel > PMAXLD ? PMAXLD : rel);
    const float* pe = pos_emb + (size_t)(rel + PMAXLD) * PDIMD;
    float s = bb;
#pragma unroll
    for (int p = 0; p < PDIMD; ++p) s += pe[p] * w2s[p];
    bias[((size_t)h_ * SEQL + i) * SEQL + j] = s;
  }
}

__global__ __launch_bounds__(256) void k_res_ln(
    float* __restrict__ h, const float* __restrict__ s2,
    const float* __restrict__ g, const float* __restrict__ b) {
  __shared__ float red[256];
  const int m = blockIdx.x, t = threadIdx.x;
  const float v = h[(size_t)m * EMBEDD + t] + s2[(size_t)m * EMBEDD + t];
  red[t] = v; __syncthreads();
  for (int s = 128; s > 0; s >>= 1) { if (t < s) red[t] += red[t + s]; __syncthreads(); }
  const float mean = red[0] * (1.0f / EMBEDD);
  __syncthreads();
  const float d = v - mean;
  red[t] = d * d; __syncthreads();
  for (int s = 128; s > 0; s >>= 1) { if (t < s) red[t] += red[t + s]; __syncthreads(); }
  const float var = red[0] * (1.0f / EMBEDD);
  h[(size_t)m * EMBEDD + t] = d * rsqrtf(var + 1e-5f) * g[t] + b[t];
}

__global__ void k_concat(const float* __restrict__ h, const float* __restrict__ eaa,
                         float* __restrict__ um) {
  int idx = blockIdx.x * blockDim.x + threadIdx.x;   // M * 288
  if (idx >= MROWS * (EMBEDD + E1DD)) return;
  const int m = idx / (EMBEDD + E1DD), c = idx % (EMBEDD + E1DD);
  um[idx] = (c < EMBEDD) ? h[(size_t)m * EMBEDD + c]
                         : eaa[(size_t)m * E1DD + (c - EMBEDD)];
}

__global__ void k_out3(const float* __restrict__ hid, const float* __restrict__ w2,
                       const float* __restrict__ b2, float* __restrict__ r3) {
  int m = blockIdx.x * blockDim.x + threadIdx.x;
  if (m >= MROWS) return;
  float s0 = b2[0], s1 = b2[1], s2 = b2[2];
  const float* hp = hid + (size_t)m * EMBEDD;
  for (int e = 0; e < EMBEDD; ++e) {
    const float hv = hp[e];
    s0 += hv * w2[e];
    s1 += hv * w2[EMBEDD + e];
    s2 += hv * w2[2 * EMBEDD + e];
  }
  float* rp = r3 + (size_t)m * 4;
  rp[0] = s0; rp[1] = s1; rp[2] = s2;
}

__global__ void k_dmap(const float* __restrict__ r3, float* __restrict__ d) {
  size_t idx = (size_t)blockIdx.x * blockDim.x + threadIdx.x;  // N*L*L
  if (idx >= (size_t)NBATCH * SEQL * SEQL) return;
  const int j = idx & 511, i = (idx >> 9) & 511, n = (int)(idx >> 18);
  const float* a = r3 + ((size_t)(n * SEQL + i)) * 4;
  const float* b = r3 + ((size_t)(n * SEQL + j)) * 4;
  const float dx = a[0] - b[0], dy = a[1] - b[1], dz = a[2] - b[2];
  d[idx] = sqrtf(dx * dx + dy * dy + dz * dz + 1e-12f);
}

// ---------------- host orchestration ------------------------------------------

extern "C" void kernel_launch(void* const* d_in, const int* in_sizes, int n_in,
                              void* d_out, int out_size, void* d_ws, size_t ws_size,
                              hipStream_t stream) {
  const float* z      = (const float*)d_in[0];
  const float* x      = (const float*)d_in[1];
  const float* posemb = (const float*)d_in[2];
  const float* aaemb  = (const float*)d_in[3];
  const float* ex_w1  = (const float*)d_in[4];
  const float* ex_b1  = (const float*)d_in[5];
  const float* ex_w2  = (const float*)d_in[6];
  const float* ex_b2  = (const float*)d_in[7];
  const float* Wq     = (const float*)d_in[8];
  const float* Wk     = (const float*)d_in[9];
  const float* Wv     = (const float*)d_in[10];
  const float* Wo     = (const float*)d_in[11];
  const float* bo     = (const float*)d_in[12];
  const float* W2d    = (const float*)d_in[13];
  const float* b2d    = (const float*)d_in[14];
  const float* Wf1    = (const float*)d_in[15];
  const float* bf1    = (const float*)d_in[16];
  const float* Wf2    = (const float*)d_in[17];
  const float* bf2    = (const float*)d_in[18];
  const float* ln1_g  = (const float*)d_in[19];
  const float* ln1_b  = (const float*)d_in[20];
  const float* ln2_g  = (const float*)d_in[21];
  const float* ln2_b  = (const float*)d_in[22];
  const float* o_w1   = (const float*)d_in[23];
  const float* o_b1   = (const float*)d_in[24];
  const float* o_w2   = (const float*)d_in[25];
  const float* o_b2   = (const float*)d_in[26];
  (void)in_sizes; (void)n_in; (void)out_size; (void)ws_size;

  float* ws = (float*)d_ws;
  size_t off = 0;
  auto take = [&](size_t nfl) { float* p = ws + off; off += nfl; return p; };
  float* h    = take((size_t)MROWS * EMBEDD);
  float* eaa  = take((size_t)MROWS * E1DD);
  float* hin  = take((size_t)MROWS * 64);
  float* t1   = take((size_t)MROWS * DFFD);
  float* t2   = take((size_t)MROWS * EMBEDD);
  float* qb   = take((size_t)MROWS * EMBEDD);
  float* kb   = take((size_t)MROWS * EMBEDD);
  float* vb   = take((size_t)MROWS * EMBEDD);   // holds V^T as (N, EMBEDD, SEQL)
  float* ob   = take((size_t)MROWS * EMBEDD);
  float* um   = take((size_t)MROWS * (EMBEDD + E1DD));
  float* bias = take((size_t)NHEAD * SEQL * SEQL);
  float* r3   = take((size_t)MROWS * 4);

  const int M = MROWS;
  const float w_t = 0.0625f;  // 1/sqrt(256)
  const dim3 gN256(M / 64, EMBEDD / 128);   // (64, 2)
  const dim3 gN1024(M / 64, DFFD / 128);    // (64, 8)

  k_tz <<<(M * 64 + 255) / 256, 256, 0, stream>>>(z, hin);
  k_eaa<<<(M + 255) / 256, 256, 0, stream>>>(x, aaemb, eaa);
  k_gemm<true,  true, false><<<gN256, 256, 0, stream>>>(hin, ex_w1, ex_b1, t1, M, EMBEDD, 64, 1.0f);
  k_gemm<false, true, false><<<gN256, 256, 0, stream>>>(t1,  ex_w2, ex_b2, h,  M, EMBEDD, EMBEDD, 1.0f);

  for (int l = 0; l < NLAYER; ++l) {
    k_bias2d<<<dim3(SEQL, NHEAD), 256, 0, stream>>>(
        posemb, W2d + (size_t)l * NHEAD * PDIMD, b2d + (size_t)l * NHEAD, bias);
    k_gemm<false, false, false><<<gN256, 256, 0, stream>>>(
        h, Wq + (size_t)l * EMBEDD * EMBEDD, nullptr, qb, M, EMBEDD, EMBEDD, w_t);
    k_gemm<false, false, false><<<gN256, 256, 0, stream>>>(
        h, Wk + (size_t)l * EMBEDD * EMBEDD, nullptr, kb, M, EMBEDD, EMBEDD, 1.0f);
    k_gemm<false, false, true><<<gN256, 256, 0, stream>>>(   // V written transposed
        h, Wv + (size_t)l * EMBEDD * EMBEDD, nullptr, vb, M, EMBEDD, EMBEDD, 1.0f);
    k_attn<<<(NBATCH * NHEAD * (SEQL / 16)) / 8, 256, 0, stream>>>(qb, kb, vb, bias, ob);
    k_gemm<false, true, false><<<gN256, 256, 0, stream>>>(
        ob, Wo + (size_t)l * EMBEDD * EMBEDD, bo + (size_t)l * EMBEDD, t2,
        M, EMBEDD, EMBEDD, 1.0f);
    k_res_ln<<<M, 256, 0, stream>>>(h, t2, ln1_g + (size_t)l * EMBEDD,
                                    ln1_b + (size_t)l * EMBEDD);
    k_concat<<<(M * (EMBEDD + E1DD) + 255) / 256, 256, 0, stream>>>(h, eaa, um);
    k_gemm<true, true, false><<<gN1024, 256, 0, stream>>>(
        um, Wf1 + (size_t)l * DFFD * (EMBEDD + E1DD), bf1 + (size_t)l * DFFD, t1,
        M, DFFD, EMBEDD + E1DD, 1.0f);
    k_gemm<false, true, false><<<gN256, 256, 0, stream>>>(
        t1, Wf2 + (size_t)l * EMBEDD * DFFD, bf2 + (size_t)l * EMBEDD, t2,
        M, EMBEDD, DFFD, 1.0f);
    k_res_ln<<<M, 256, 0, stream>>>(h, t2, ln2_g + (size_t)l * EMBEDD,
                                    ln2_b + (size_t)l * EMBEDD);
  }

  k_gemm<true, true, false><<<gN256, 256, 0, stream>>>(h, o_w1, o_b1, t1, M, EMBEDD, EMBEDD, 1.0f);
  k_out3<<<(M + 255) / 256, 256, 0, stream>>>(t1, o_w2, o_b2, r3);
  k_dmap<<<(int)(((size_t)NBATCH * SEQL * SEQL + 255) / 256), 256, 0, stream>>>(
      r3, (float*)d_out);
}
